// KalmanNetNN_14508399526739
// MI455X (gfx1250) — matrix operand order repaired and checked by reading, not compiled, not run on hardware
//
#include <hip/hip_runtime.h>
#include <math.h>

// KalmanNet persistent-kernel for gfx1250 (MI455X), round 3.
// - bf16 weight tiles (L2-resident, pre-swizzled for v_wmma_f32_16x16x32_bf16)
// - 3 grid barriers per recurrent step; small math redundant per block.
// - Activations in LDS; WMMA B operand via ds_load_b128 (LDS broadcast).
// - gemv inner loop explicitly software-pipelined (prefetch depth 3) and
//   unit indices scalarized via readfirstlane (saddr loads, scalar loops).

typedef __attribute__((ext_vector_type(16))) __bf16 v16bf;
typedef __attribute__((ext_vector_type(8)))  float  v8f;

#define MM    10          // state dim m
#define NNV   10          // obs dim n
#define TT    100         // time steps
#define DIN   30          // 2m+n
#define H1N   1600
#define HID   2000
#define KP    2016        // HID padded to multiple of 32
#define H2N   400
#define G3    6000        // 3*HID
#define RT_G  375         // 6000/16 row tiles
#define RT_W2 25          // 400/16
#define KT_H1 50          // 1600/32
#define KT_HID 63         // 2016/32

#define NBLK  96
#define TPB   512
#define NWAVES (NBLK * (TPB / 32))

__device__ __forceinline__ __bf16 to_bf16(float f) {
  unsigned u = __float_as_uint(f);
  unsigned r = u + 0x7FFFu + ((u >> 16) & 1u);      // round-to-nearest-even
  unsigned short h = (unsigned short)(r >> 16);
  __bf16 b;
  __builtin_memcpy(&b, &h, 2);
  return b;
}

__device__ __forceinline__ float sigm(float x) {
  return 1.0f / (1.0f + __expf(-x));
}

// Grid-wide generation barrier (agent scope). Fences make producer stores
// visible through L2; tight poll minimizes release latency.
__device__ __forceinline__ void gridBarrier(unsigned* cnt, unsigned* gen, unsigned nb) {
  __threadfence();
  __syncthreads();
  if (threadIdx.x == 0) {
    unsigned g = __hip_atomic_load(gen, __ATOMIC_RELAXED, __HIP_MEMORY_SCOPE_AGENT);
    unsigned old = __hip_atomic_fetch_add(cnt, 1u, __ATOMIC_ACQ_REL, __HIP_MEMORY_SCOPE_AGENT);
    if (old == nb - 1u) {
      __hip_atomic_store(cnt, 0u, __ATOMIC_RELAXED, __HIP_MEMORY_SCOPE_AGENT);
      __hip_atomic_fetch_add(gen, 1u, __ATOMIC_RELEASE, __HIP_MEMORY_SCOPE_AGENT);
    } else {
      while (__hip_atomic_load(gen, __ATOMIC_RELAXED, __HIP_MEMORY_SCOPE_AGENT) == g) {
      }
    }
  }
  __syncthreads();
  __threadfence();
}

// One GEMV work unit: 16 output rows (row tile rt), K tiles [kt0,kt1).
// Explicit software pipeline, prefetch depth 3: loads for tile k+3 are issued
// before the WMMA for tile k, so 6+ loads stay in flight and the pre-WMMA
// s_wait leaves prefetches outstanding. Requires kt1-kt0 >= 3 (min here: 15).
// rt/kt0/kt1 must be wave-uniform (callers pass readfirstlane'd values).
__device__ __forceinline__ void gemv_unit(const __bf16* __restrict__ tiles, int kTilesTotal,
                                          const __bf16* xlds,   // LDS-resident vector
                                          float* __restrict__ outv,
                                          int rt, int kt0, int kt1, int lane) {
  const int n = kt1 - kt0;
  v8f c = {};
  const __bf16* tp = tiles + ((size_t)rt * kTilesTotal + kt0) * 512 + lane * 16;
  const __bf16* xp = xlds + kt0 * 32 + ((lane & 16) ? 16 : 0);

  v16bf a0 = *(const v16bf*)(tp);
  v16bf b0 = *(const v16bf*)(xp);
  v16bf a1 = *(const v16bf*)(tp + 512);
  v16bf b1 = *(const v16bf*)(xp + 32);
  v16bf a2 = *(const v16bf*)(tp + 1024);
  v16bf b2 = *(const v16bf*)(xp + 64);
  #pragma unroll 3
  for (int i = 0; i < n - 3; ++i) {
    v16bf a3 = *(const v16bf*)(tp + 1536);
    v16bf b3 = *(const v16bf*)(xp + 96);
    c = __builtin_amdgcn_wmma_f32_16x16x32_bf16(false, a0, false, b0, (short)0, c, false, false);
    a0 = a1; b0 = b1; a1 = a2; b1 = b2; a2 = a3; b2 = b3;
    tp += 512; xp += 32;
  }
  c = __builtin_amdgcn_wmma_f32_16x16x32_bf16(false, a0, false, b0, (short)0, c, false, false);
  c = __builtin_amdgcn_wmma_f32_16x16x32_bf16(false, a1, false, b1, (short)0, c, false, false);
  c = __builtin_amdgcn_wmma_f32_16x16x32_bf16(false, a2, false, b2, (short)0, c, false, false);

  int r0 = rt * 16;
  if (lane == 0) {
    #pragma unroll
    for (int r = 0; r < 8; ++r) outv[r0 + r] = c[r];
  } else if (lane == 16) {
    #pragma unroll
    for (int r = 0; r < 8; ++r) outv[r0 + 8 + r] = c[r];
  }
}

// GRU gate combine (per-block redundant): gip holds `nparts` K-split partials
// of Wi@x (each G3 long), gh the full Wh@h. State hf/hbf in this block's LDS.
template <int NPARTS>
__device__ __forceinline__ void gru_combine_lds(const float* __restrict__ gip,
                                                const float* __restrict__ gh,
                                                const float* __restrict__ bi,
                                                const float* __restrict__ bh,
                                                float* hf, __bf16* hbf, int tid) {
  for (int j = tid; j < HID; j += TPB) {
    float xr = bi[j] + gh[j] + bh[j];
    float xz = bi[HID + j] + gh[HID + j] + bh[HID + j];
    float xn = bi[2 * HID + j];
    float hn = gh[2 * HID + j] + bh[2 * HID + j];
    #pragma unroll
    for (int p = 0; p < NPARTS; ++p) {
      xr += gip[p * G3 + j];
      xz += gip[p * G3 + HID + j];
      xn += gip[p * G3 + 2 * HID + j];
    }
    float r = sigm(xr);
    float z = sigm(xz);
    float nn2 = tanhf(xn + r * hn);
    float h = (1.0f - z) * nn2 + z * hf[j];
    hf[j] = h;
    hbf[j] = to_bf16(h);
  }
}

// f32 -> bf16 weight conversion + swizzle into WMMA A-operand tile layout.
__global__ __launch_bounds__(256) void knet_swizzle(const float* __restrict__ src,
                                                    __bf16* __restrict__ dst,
                                                    int rowTiles, int K, int kTiles) {
  size_t total = (size_t)rowTiles * kTiles * 512;
  for (size_t idx = (size_t)blockIdx.x * blockDim.x + threadIdx.x; idx < total;
       idx += (size_t)gridDim.x * blockDim.x) {
    int j    = (int)(idx & 15);
    int lane = (int)((idx >> 4) & 31);
    size_t tf = idx >> 9;
    int kt = (int)(tf % kTiles);
    int rt = (int)(tf / kTiles);
    int row = rt * 16 + (lane & 15);
    int koff = (lane < 16) ? j : j + 16;
    int k = kt * 32 + koff;
    float v = (k < K) ? src[(size_t)row * K + k] : 0.0f;
    dst[idx] = to_bf16(v);
  }
}

__global__ __launch_bounds__(TPB) void knet_persistent(
    const float* __restrict__ y, const float* __restrict__ m1x0,
    const float* __restrict__ Fm, const float* __restrict__ Hm,
    const float* __restrict__ h0,
    const float* __restrict__ W1, const float* __restrict__ b1,
    const float* __restrict__ bi0, const float* __restrict__ bh0,
    const float* __restrict__ bi1, const float* __restrict__ bh1,
    const float* __restrict__ b2, const float* __restrict__ W3,
    const float* __restrict__ b3,
    const __bf16* __restrict__ Wi0t, const __bf16* __restrict__ Wh0t,
    const __bf16* __restrict__ Wi1t, const __bf16* __restrict__ Wh1t,
    const __bf16* __restrict__ W2t,
    float* __restrict__ gi0p, float* __restrict__ gh0,
    float* __restrict__ gi1p, float* __restrict__ gh1,
    float* __restrict__ grawp,
    unsigned* barCnt, unsigned* barGen,
    float* __restrict__ out) {
  // Per-block (per-WGP) LDS-resident activations and state.
  __shared__ __attribute__((aligned(32))) __bf16 l_h1bf[KP];
  __shared__ __attribute__((aligned(32))) __bf16 l_h2bf[KP];
  __shared__ __attribute__((aligned(32))) __bf16 l_abf[H1N];
  __shared__ float l_h1f[HID];
  __shared__ float l_h2f[HID];
  __shared__ float s_post[MM], s_prev_post[MM], s_prev_prior[MM], s_yprev[NNV];
  __shared__ float s_prior[MM], s_m1y[NNV], s_tmp[MM], s_kin[DIN], s_norm[3];
  __shared__ float s_gv[H2N], s_KG[MM * NNV], s_dy[NNV];

  const int tid  = threadIdx.x;
  const int lane = tid & 31;
  // Wave index is uniform across the wave: pin it to an SGPR so unit loops,
  // tile indices and K ranges become scalar (s_cbranch loops, saddr loads).
  const int wvs  = __builtin_amdgcn_readfirstlane(blockIdx.x * (TPB / 32) + (tid >> 5));

  // ---------------- per-block init (no grid barrier needed) ----------------
  for (int j = tid; j < HID; j += TPB) {
    float v1 = h0[j], v2 = h0[HID + j];
    l_h1f[j] = v1; l_h2f[j] = v2;
    l_h1bf[j] = to_bf16(v1); l_h2bf[j] = to_bf16(v2);
  }
  if (tid < KP - HID) {
    l_h1bf[HID + tid] = to_bf16(0.0f);
    l_h2bf[HID + tid] = to_bf16(0.0f);
  }
  if (tid < MM) {
    float x0 = m1x0[tid];
    s_post[tid] = x0; s_prev_post[tid] = 0.0f; s_prev_prior[tid] = x0;
  }
  __syncthreads();
  if (tid < MM) {
    float s = 0.0f;
    for (int j = 0; j < MM; ++j) s += Fm[tid * MM + j] * s_post[j];
    s_tmp[tid] = s;
  }
  __syncthreads();
  if (tid < NNV) {
    float s = 0.0f;
    for (int j = 0; j < MM; ++j) s += Hm[tid * MM + j] * s_tmp[j];
    s_yprev[tid] = s;
  }
  __syncthreads();

  for (int t = 0; t <= TT; ++t) {
    // ======== redundant G(t-1): g, KG, posterior update, output ========
    if (t > 0) {
      for (int i2 = tid; i2 < H2N; i2 += TPB)
        s_gv[i2] = fmaxf(grawp[i2] + grawp[H2N + i2] + b2[i2], 0.0f);
      if (tid < NNV) s_dy[tid] = y[tid * TT + (t - 1)] - s_m1y[tid];
      __syncthreads();
      for (int r = tid; r < MM * NNV; r += TPB) {
        float s = b3[r];
        for (int c2 = 0; c2 < H2N; ++c2) s += W3[r * H2N + c2] * s_gv[c2];
        s_KG[r] = s;
      }
      __syncthreads();
      if (tid < MM) {
        float np = s_prior[tid];
        for (int j = 0; j < NNV; ++j) np += s_KG[tid * NNV + j] * s_dy[j];
        if (blockIdx.x == 0) out[tid * TT + (t - 1)] = np;
        s_prev_post[tid]  = s_post[tid];
        s_prev_prior[tid] = s_prior[tid];
        s_post[tid]       = np;
        s_yprev[tid]      = y[tid * TT + (t - 1)];
      }
      __syncthreads();
    }
    if (t == TT) break;

    // ======== redundant small(t): prior, features, kin, a = relu(W1 kin) ====
    if (tid < MM) {
      float s = 0.0f;
      for (int j = 0; j < MM; ++j) s += Fm[tid * MM + j] * s_post[j];
      s_prior[tid] = s;
    }
    __syncthreads();
    if (tid < NNV) {
      float s = 0.0f;
      for (int j = 0; j < MM; ++j) s += Hm[tid * MM + j] * s_prior[j];
      s_m1y[tid] = s;
    }
    if (tid == 0) {
      float n1 = 0.0f, n3 = 0.0f, n4 = 0.0f;
      for (int j = 0; j < NNV; ++j) { float d = y[j * TT + t] - s_yprev[j]; n1 += d * d; }
      for (int j = 0; j < MM; ++j) {
        float d = s_post[j] - s_prev_post[j];  n3 += d * d;
        float e = s_post[j] - s_prev_prior[j]; n4 += e * e;
      }
      s_norm[0] = fmaxf(sqrtf(n1), 1e-12f);
      s_norm[1] = fmaxf(sqrtf(n3), 1e-12f);
      s_norm[2] = fmaxf(sqrtf(n4), 1e-12f);
    }
    __syncthreads();
    if (tid < NNV) s_kin[tid] = (y[tid * TT + t] - s_yprev[tid]) / s_norm[0];
    if (tid < MM) {
      s_kin[NNV + tid]      = (s_post[tid] - s_prev_post[tid])  / s_norm[1];
      s_kin[NNV + MM + tid] = (s_post[tid] - s_prev_prior[tid]) / s_norm[2];
    }
    __syncthreads();
    for (int r = tid; r < H1N; r += TPB) {
      float s = b1[r];
      #pragma unroll
      for (int j = 0; j < DIN; ++j) s += W1[r * DIN + j] * s_kin[j];
      l_abf[r] = to_bf16(fmaxf(s, 0.0f));
    }
    __syncthreads();

    // ======== Phase alpha (distributed WMMA): gh0, gh1, gi0 ========
    // units: [0,375) gh0 | [375,750) gh1 | [750,1500) gi0 (K split 2-way)
    for (int u = wvs; u < 4 * RT_G; u += NWAVES) {
      if (u < RT_G) {
        gemv_unit(Wh0t, KT_HID, l_h1bf, gh0, u, 0, KT_HID, lane);
      } else if (u < 2 * RT_G) {
        gemv_unit(Wh1t, KT_HID, l_h2bf, gh1, u - RT_G, 0, KT_HID, lane);
      } else {
        int v2 = u - 2 * RT_G;
        int rt = v2 >> 1, half = v2 & 1;
        gemv_unit(Wi0t, KT_H1, l_abf, gi0p + half * G3, rt,
                  half ? 25 : 0, half ? KT_H1 : 25, lane);
      }
    }
    gridBarrier(barCnt, barGen, NBLK);

    // ======== Phase gamma: combine h1 (redundant) + gi1 (distributed) ======
    gru_combine_lds<2>(gi0p, gh0, bi0, bh0, l_h1f, l_h1bf, tid);
    __syncthreads();
    // gi1: 1500 units, K split 4-way (16,16,16,15 tiles) to feed all waves
    for (int u = wvs; u < 4 * RT_G; u += NWAVES) {
      int rt = u >> 2, q = u & 3;
      int kt0 = q * 16;
      int kt1 = (q == 3) ? KT_HID : kt0 + 16;
      gemv_unit(Wi1t, KT_HID, l_h1bf, gi1p + q * G3, rt, kt0, kt1, lane);
    }
    gridBarrier(barCnt, barGen, NBLK);

    // ======== Phase delta: combine h2 (redundant) + W2 (distributed) =======
    gru_combine_lds<4>(gi1p, gh1, bi1, bh1, l_h2f, l_h2bf, tid);
    __syncthreads();
    for (int u = wvs; u < 2 * RT_W2; u += NWAVES) {
      int rt = u >> 1, half = u & 1;
      gemv_unit(W2t, KT_HID, l_h2bf, grawp + half * H2N, rt,
                half ? 32 : 0, half ? KT_HID : 32, lane);
    }
    gridBarrier(barCnt, barGen, NBLK);
  }
}

extern "C" void kernel_launch(void* const* d_in, const int* in_sizes, int n_in,
                              void* d_out, int out_size, void* d_ws, size_t ws_size,
                              hipStream_t stream) {
  (void)in_sizes; (void)n_in; (void)out_size;
  const float* y    = (const float*)d_in[0];
  const float* m1x0 = (const float*)d_in[1];
  const float* Fm   = (const float*)d_in[2];
  const float* Hm   = (const float*)d_in[3];
  const float* h0   = (const float*)d_in[4];
  const float* W1   = (const float*)d_in[5];
  const float* b1   = (const float*)d_in[6];
  const float* Wi0  = (const float*)d_in[7];
  const float* Wh0  = (const float*)d_in[8];
  const float* bi0  = (const float*)d_in[9];
  const float* bh0  = (const float*)d_in[10];
  const float* Wi1  = (const float*)d_in[11];
  const float* Wh1  = (const float*)d_in[12];
  const float* bi1  = (const float*)d_in[13];
  const float* bh1  = (const float*)d_in[14];
  const float* W2   = (const float*)d_in[15];
  const float* b2   = (const float*)d_in[16];
  const float* W3   = (const float*)d_in[17];
  const float* b3   = (const float*)d_in[18];
  float* out = (float*)d_out;

  size_t off = 0;
  char* base = (char*)d_ws;
  auto take = [&](size_t bytes) -> void* {
    void* p = base + off;
    off += (bytes + 255) & ~(size_t)255;
    return p;
  };
  unsigned* bar  = (unsigned*)take(256);
  __bf16* Wi0t = (__bf16*)take((size_t)RT_G  * KT_H1  * 512 * 2);
  __bf16* Wh0t = (__bf16*)take((size_t)RT_G  * KT_HID * 512 * 2);
  __bf16* Wi1t = (__bf16*)take((size_t)RT_G  * KT_HID * 512 * 2);
  __bf16* Wh1t = (__bf16*)take((size_t)RT_G  * KT_HID * 512 * 2);
  __bf16* W2t  = (__bf16*)take((size_t)RT_W2 * KT_HID * 512 * 2);
  float*  gi0p = (float*)take((size_t)2 * G3 * 4);
  float*  gh0  = (float*)take((size_t)G3 * 4);
  float*  gi1p = (float*)take((size_t)4 * G3 * 4);
  float*  gh1  = (float*)take((size_t)G3 * 4);
  float*  graw = (float*)take((size_t)2 * H2N * 4);
  if (off > ws_size) return;  // workspace too small; nothing safe to do

  hipMemsetAsync(bar, 0, 256, stream);

  // Weight conversion + WMMA tile swizzle (once per launch; ~190MB read).
  knet_swizzle<<<3072, 256, 0, stream>>>(Wi0, Wi0t, RT_G, H1N, KT_H1);
  knet_swizzle<<<3072, 256, 0, stream>>>(Wh0, Wh0t, RT_G, HID, KT_HID);
  knet_swizzle<<<3072, 256, 0, stream>>>(Wi1, Wi1t, RT_G, HID, KT_HID);
  knet_swizzle<<<3072, 256, 0, stream>>>(Wh1, Wh1t, RT_G, HID, KT_HID);
  knet_swizzle<<<1024, 256, 0, stream>>>(W2,  W2t,  RT_W2, HID, KT_HID);

  knet_persistent<<<NBLK, TPB, 0, stream>>>(
      y, m1x0, Fm, Hm, h0, W1, b1, bi0, bh0, bi1, bh1, b2, W3, b3,
      Wi0t, Wh0t, Wi1t, Wh1t, W2t,
      gi0p, gh0, gi1p, gh1, graw,
      bar, bar + 1, out);
}
